// S2S_Model_12713103196187
// MI455X (gfx1250) — matrix operand order repaired and checked by reading, not compiled, not run on hardware
//
#include <hip/hip_runtime.h>
#include <hip/hip_bf16.h>

#define BATCH 256
#define HID   512
#define INSZ  64
#define SRCL  512
#define TGTL  96
#define NB    32      // HID/16 blocks, each owns a 16-wide hidden slice
#define NT    256     // 8 wave32 per block

typedef __bf16 bf16_t;
typedef __attribute__((ext_vector_type(16))) __bf16 v16bf;
typedef __attribute__((ext_vector_type(8)))  __bf16 v8bf;
typedef __attribute__((ext_vector_type(8)))  float  v8f;
typedef __attribute__((ext_vector_type(4)))  float  v4f;

__device__ inline v8f wmma_bf16(v16bf a, v16bf b, v8f c) {
  // v_wmma_f32_16x16x32_bf16  (neg_a, A, neg_b, B, c_mod, C, reuse_a, reuse_b)
  return __builtin_amdgcn_wmma_f32_16x16x32_bf16(false, a, false, b, (short)0, c, false, false);
}

// A fragment (16x32 bf16, MxK): lane holds K = {kb..kb+7, kb+16..kb+23}; two 16B chunks.
__device__ inline v16bf load_a_bf16(const bf16_t* p) {
  v8bf lo = *(const v8bf*)(p);
  v8bf hi = *(const v8bf*)(p + 16);
  v16bf r;
#pragma unroll
  for (int i = 0; i < 8; ++i) { r[i] = lo[i]; r[8 + i] = hi[i]; }
  return r;
}

// A fragment built from fp32 source, converted to bf16 on the fly (fallback path).
__device__ inline v16bf load_a_f32(const float* p) {
  v4f a0 = *(const v4f*)(p);
  v4f a1 = *(const v4f*)(p + 4);
  v4f b0 = *(const v4f*)(p + 16);
  v4f b1 = *(const v4f*)(p + 20);
  v16bf r;
#pragma unroll
  for (int i = 0; i < 4; ++i) {
    r[i]      = (bf16_t)a0[i];
    r[4 + i]  = (bf16_t)a1[i];
    r[8 + i]  = (bf16_t)b0[i];
    r[12 + i] = (bf16_t)b1[i];
  }
  return r;
}

// B fragment (32x16 bf16, KxN): lane holds K = {kb..kb+7, kb+16..kb+23}; two 16B chunks.
__device__ inline v16bf load_b_lds(const bf16_t* p) {
  v8bf lo = *(const v8bf*)(p);
  v8bf hi = *(const v8bf*)(p + 16);
  v16bf r;
#pragma unroll
  for (int i = 0; i < 8; ++i) { r[i] = lo[i]; r[8 + i] = hi[i]; }
  return r;
}

__device__ inline v8f splat8(float f) {
  v8f r;
#pragma unroll
  for (int i = 0; i < 8; ++i) r[i] = f;
  return r;
}

__device__ inline float sigmoidf_(float x) { return 1.0f / (1.0f + __expf(-x)); }

// One-time x (fp32) -> bf16 conversion: removes per-step cvt chains from the
// recurrent hot loop (x is read 512 times, converted once).
__global__ void __launch_bounds__(NT)
xcvt_kernel(const float* __restrict__ x, bf16_t* __restrict__ xbf, int n) {
  int i = (blockIdx.x * NT + threadIdx.x) * 8;
  if (i + 7 < n) {
    v4f a = *(const v4f*)(x + i);
    v4f b = *(const v4f*)(x + i + 4);
    v8bf o;
#pragma unroll
    for (int j = 0; j < 4; ++j) { o[j] = (bf16_t)a[j]; o[4 + j] = (bf16_t)b[j]; }
    *(v8bf*)(xbf + i) = o;
  }
}

template <bool USE_XBF>
__global__ void __launch_bounds__(NT, 1)
s2s_gru_persistent(const float* __restrict__ x,
                   const bf16_t* __restrict__ xbf,   // pre-converted x (when USE_XBF)
                   const float* __restrict__ WihE, const float* __restrict__ WhhE,
                   const float* __restrict__ bihE, const float* __restrict__ bhhE,
                   const float* __restrict__ WihD, const float* __restrict__ WhhD,
                   const float* __restrict__ bihD, const float* __restrict__ bhhD,
                   const float* __restrict__ Wlin, const float* __restrict__ blin,
                   unsigned* __restrict__ bar,
                   bf16_t* __restrict__ hbf,     // 2 x [BATCH][HID] ping-pong
                   float* __restrict__ out)      // [BATCH][TGTL]
{
  __shared__ bf16_t sWhh[3][16][HID];   // 48 KB: B operand (rows of W_hh slice)
  __shared__ bf16_t sWih[3][16][INSZ];  //  6 KB: encoder input weights
  __shared__ float  sBih[3][16];
  __shared__ float  sBhh[3][16];
  __shared__ float  sWid[3][16];        // decoder W_ih (column vector slice)
  __shared__ float  sWlin[HID];         //  2 KB

  const int tid  = threadIdx.x;
  const int wave = tid >> 5;
  const int lane = tid & 31;
  const int n16  = lane & 15;
  const int hi   = lane >> 4;           // 0/1: which M-half of C/D this lane holds
  const int kbA  = (hi == 0) ? 0 : 8;   // A-fragment K base
  const int kbB  = (hi == 0) ? 0 : 8;   // B-fragment K base
  const int blk  = blockIdx.x;
  const int c0   = blk * 16;            // hidden-slice start column
  const int mt0  = wave * 2;            // this wave's first M tile (2 tiles of 16 rows)

  // ---- stage encoder weights into LDS (bf16) ----
  for (int i = tid; i < 3 * 16 * HID; i += NT) {
    int g = i / (16 * HID); int rem = i - g * 16 * HID;
    int n = rem / HID;      int k   = rem % HID;
    sWhh[g][n][k] = (bf16_t)WhhE[(size_t)(g * HID + c0 + n) * HID + k];
  }
  for (int i = tid; i < 3 * 16 * INSZ; i += NT) {
    int g = i / (16 * INSZ); int rem = i - g * 16 * INSZ;
    int n = rem / INSZ;      int k   = rem % INSZ;
    sWih[g][n][k] = (bf16_t)WihE[(size_t)(g * HID + c0 + n) * INSZ + k];
  }
  if (tid < 48) {
    int g = tid / 16, n = tid % 16;
    sBih[g][n] = bihE[g * HID + c0 + n];
    sBhh[g][n] = bhhE[g * HID + c0 + n];
  }
  for (int i = tid; i < HID; i += NT) sWlin[i] = Wlin[i];
  __syncthreads();

  const float bl = blin[0];

  // fp32 hidden state held in registers: element (mt, v) of this lane is
  // h[b = (mt0+mt)*16 + v + 8*hi][c0 + n16]
  float hreg[2][8];
#pragma unroll
  for (int mt = 0; mt < 2; ++mt)
#pragma unroll
    for (int v = 0; v < 8; ++v) hreg[mt][v] = 0.0f;

  unsigned gen = 0;
  int cur = 0;

  for (int phase = 0; phase < 2; ++phase) {
    const int nsteps = (phase == 0) ? SRCL : TGTL;
    if (phase == 1) {
      __syncthreads();
      for (int i = tid; i < 3 * 16 * HID; i += NT) {
        int g = i / (16 * HID); int rem = i - g * 16 * HID;
        int n = rem / HID;      int k   = rem % HID;
        sWhh[g][n][k] = (bf16_t)WhhD[(size_t)(g * HID + c0 + n) * HID + k];
      }
      if (tid < 48) {
        int g = tid / 16, n = tid % 16;
        sBih[g][n] = bihD[g * HID + c0 + n];
        sBhh[g][n] = bhhD[g * HID + c0 + n];
        sWid[g][n] = WihD[g * HID + c0 + n];   // W_ih_d is (3H, 1)
      }
      __syncthreads();
    }

    for (int t = 0; t < nsteps; ++t) {
      const bf16_t* hsrc = hbf + (size_t)cur * BATCH * HID;

      // ---- decoder input: pred[b] = h_prev . Wlin (deterministic, per wave) ----
      float praw = 0.0f;
      if (phase == 1) {
        const int row = mt0 * 16 + lane;            // wave covers its own 32 rows
        const bf16_t* hp = hsrc + (size_t)row * HID;
        for (int k = 0; k < HID; k += 8) {
          v8bf hv = *(const v8bf*)(hp + k);
#pragma unroll
          for (int j = 0; j < 8; ++j) praw += (float)hv[j] * sWlin[k + j];
        }
        if (blk == 0 && t > 0) out[(size_t)row * TGTL + (t - 1)] = praw + bl;
      }

      // ---- init accumulators (biases, plus decoder gi outer-product) ----
      v8f acc[2][4];   // [mt][r, z, gi_n, gh_n]
      const float b0 = sBih[0][n16] + sBhh[0][n16];
      const float b1 = sBih[1][n16] + sBhh[1][n16];
      const float b2i = sBih[2][n16];
      const float b2h = sBhh[2][n16];
      if (phase == 0) {
#pragma unroll
        for (int mt = 0; mt < 2; ++mt) {
          acc[mt][0] = splat8(b0);
          acc[mt][1] = splat8(b1);
          acc[mt][2] = splat8(b2i);
          acc[mt][3] = splat8(b2h);
        }
      } else {
        const float w0 = sWid[0][n16], w1 = sWid[1][n16], w2 = sWid[2][n16];
#pragma unroll
        for (int mt = 0; mt < 2; ++mt) {
#pragma unroll
          for (int v = 0; v < 8; ++v) {
            float pin = __shfl(praw, mt * 16 + 8 * hi + v, 32) + bl;
            acc[mt][0][v] = b0 + pin * w0;
            acc[mt][1][v] = b1 + pin * w1;
            acc[mt][2][v] = b2i + pin * w2;
            acc[mt][3][v] = b2h;
          }
        }
      }

      // ---- gi GEMM (encoder only): x_t @ W_ih_e^T, K = 64 ----
      if (phase == 0) {
#pragma unroll
        for (int k0 = 0; k0 < INSZ; k0 += 32) {
          v16bf bf0 = load_b_lds(&sWih[0][n16][k0 + kbB]);
          v16bf bf1 = load_b_lds(&sWih[1][n16][k0 + kbB]);
          v16bf bf2 = load_b_lds(&sWih[2][n16][k0 + kbB]);
#pragma unroll
          for (int mt = 0; mt < 2; ++mt) {
            const size_t row = (size_t)((mt0 + mt) * 16 + n16);
            v16bf af;
            if constexpr (USE_XBF) {
              af = load_a_bf16(xbf + (row * SRCL + t) * INSZ + k0 + kbA);
            } else {
              af = load_a_f32(x + (row * SRCL + t) * INSZ + k0 + kbA);
            }
            acc[mt][0] = wmma_bf16(af, bf0, acc[mt][0]);
            acc[mt][1] = wmma_bf16(af, bf1, acc[mt][1]);
            acc[mt][2] = wmma_bf16(af, bf2, acc[mt][2]);
          }
        }
      }

      // ---- gh GEMM: h @ W_hh^T, K = 512 ----
#pragma unroll 2
      for (int k0 = 0; k0 < HID; k0 += 32) {
        v16bf bf0 = load_b_lds(&sWhh[0][n16][k0 + kbB]);
        v16bf bf1 = load_b_lds(&sWhh[1][n16][k0 + kbB]);
        v16bf bf2 = load_b_lds(&sWhh[2][n16][k0 + kbB]);
#pragma unroll
        for (int mt = 0; mt < 2; ++mt) {
          const size_t row = (size_t)((mt0 + mt) * 16 + n16);
          const bf16_t* ap = hsrc + row * HID + k0 + kbA;
          if (k0 + 64 < HID) __builtin_prefetch(ap + 64, 0, 0);  // next K chunk (L2-resident h)
          v16bf af = load_a_bf16(ap);
          acc[mt][0] = wmma_bf16(af, bf0, acc[mt][0]);
          acc[mt][1] = wmma_bf16(af, bf1, acc[mt][1]);
          acc[mt][3] = wmma_bf16(af, bf2, acc[mt][3]);
        }
      }

      // ---- GRU elementwise update (block-local hidden slice) ----
      bf16_t* hdst = hbf + (size_t)(cur ^ 1) * BATCH * HID;
#pragma unroll
      for (int mt = 0; mt < 2; ++mt) {
#pragma unroll
        for (int v = 0; v < 8; ++v) {
          float r = sigmoidf_(acc[mt][0][v]);
          float z = sigmoidf_(acc[mt][1][v]);
          float n = tanhf(acc[mt][2][v] + r * acc[mt][3][v]);
          float hn = (1.0f - z) * n + z * hreg[mt][v];
          hreg[mt][v] = hn;
          int b = (mt0 + mt) * 16 + v + 8 * hi;
          hdst[(size_t)b * HID + c0 + n16] = (bf16_t)hn;
        }
      }

      // ---- device-wide step barrier ----
      __threadfence();
      __syncthreads();
      ++gen;
      if (tid == 0) {
        atomicAdd(bar, 1u);
        volatile unsigned* vb = bar;
        while (*vb < gen * (unsigned)NB) __builtin_amdgcn_s_sleep(1);
      }
      __syncthreads();
      __threadfence();
      cur ^= 1;
    }
  }

  // ---- final prediction (after last decoder update) ----
  if (blk == 0) {
    const int row = mt0 * 16 + lane;
    const bf16_t* hp = hbf + (size_t)cur * BATCH * HID + (size_t)row * HID;
    float praw = 0.0f;
    for (int k = 0; k < HID; k += 8) {
      v8bf hv = *(const v8bf*)(hp + k);
#pragma unroll
      for (int j = 0; j < 8; ++j) praw += (float)hv[j] * sWlin[k + j];
    }
    out[(size_t)row * TGTL + (TGTL - 1)] = praw + bl;
  }
}

extern "C" void kernel_launch(void* const* d_in, const int* in_sizes, int n_in,
                              void* d_out, int out_size, void* d_ws, size_t ws_size,
                              hipStream_t stream) {
  (void)in_sizes; (void)n_in; (void)out_size;
  const float* x    = (const float*)d_in[0];
  const float* WihE = (const float*)d_in[1];
  const float* WhhE = (const float*)d_in[2];
  const float* bihE = (const float*)d_in[3];
  const float* bhhE = (const float*)d_in[4];
  const float* WihD = (const float*)d_in[5];
  const float* WhhD = (const float*)d_in[6];
  const float* bihD = (const float*)d_in[7];
  const float* bhhD = (const float*)d_in[8];
  const float* Wlin = (const float*)d_in[9];
  const float* blin = (const float*)d_in[10];

  unsigned char* ws = (unsigned char*)d_ws;
  unsigned* bar = (unsigned*)ws;                       // 4B counter (256B pad)
  bf16_t*  hbf = (bf16_t*)(ws + 256);                  // 2 x 256KB ping-pong
  const size_t hbf_bytes = 2 * (size_t)BATCH * HID * sizeof(bf16_t);
  const size_t xbf_off   = 256 + hbf_bytes;
  const size_t xn        = (size_t)BATCH * SRCL * INSZ;      // 8.39M elements
  const size_t need_xbf  = xbf_off + xn * sizeof(bf16_t);

  hipMemsetAsync(d_ws, 0, 256 + hbf_bytes, stream);    // h0 = 0, barrier gen = 0

  if (ws_size >= need_xbf) {                           // ws_size fixed per run -> deterministic
    bf16_t* xbf = (bf16_t*)(ws + xbf_off);
    int nblk = (int)(xn / 8 + NT - 1) / NT;
    xcvt_kernel<<<nblk, NT, 0, stream>>>(x, xbf, (int)xn);
    s2s_gru_persistent<true><<<NB, NT, 0, stream>>>(
        x, xbf, WihE, WhhE, bihE, bhhE, WihD, WhhD, bihD, bhhD, Wlin, blin,
        bar, hbf, (float*)d_out);
  } else {
    s2s_gru_persistent<false><<<NB, NT, 0, stream>>>(
        x, nullptr, WihE, WhhE, bihE, bhhE, WihD, WhhD, bihD, bhhD, Wlin, blin,
        bar, hbf, (float*)d_out);
  }
}
